// Attention_58059367908033
// MI455X (gfx1250) — matrix operand order repaired and checked by reading, not compiled, not run on hardware
//
#include <hip/hip_runtime.h>

// ---------------------------------------------------------------------------
// Dual-key attention for MI455X (gfx1250), bf16 WMMA, flash-attention tiling.
//   fmap [2,256,64,64] f32, x [2,2048,64,64] f32, Wqkv [768,256], Wk2 [256,2048]
//   out  [2,256,64,64] f32
// ws layout (bf16, [b,h,n,d] each = 2*2*4096*128 elems = 4 MB):
//   q (pre-scaled by 128^-0.5) | ksum (= k1 + k2) | v
// ---------------------------------------------------------------------------

typedef __bf16 bf16;
typedef __bf16 v16bf __attribute__((ext_vector_type(16)));
typedef __bf16 v8bf  __attribute__((ext_vector_type(8)));
typedef float  v8f   __attribute__((ext_vector_type(8)));

// Types matching the async-LDS builtin signature exactly (from hipcc diagnostic):
//   param0: __attribute__((vector_size(16))) int __device__ *   (addrspace(1))
//   param1: LDS destination (addrspace(3))
typedef int v4i_vs __attribute__((vector_size(16)));
typedef __attribute__((address_space(1))) v4i_vs* as1_v4i;
typedef __attribute__((address_space(3))) v4i_vs* as3_v4i;

#define N_SPATIAL 4096
#define DHEAD     128
#define BH        4      // batch * heads

#if defined(__gfx1250__) && __has_builtin(__builtin_amdgcn_global_load_async_to_lds_b128)
#define HAVE_ASYNC_LDS 1
#else
#define HAVE_ASYNC_LDS 0
#endif

union BF8Pack { uint4 u; bf16 b[8]; };

// Contiguous 16 bf16 (B-matrix fragment: lanes 0-15 K=0..15, lanes 16-31 K=16..31;
// caller adds the half*16 offset).
__device__ __forceinline__ v16bf ld16(const bf16* p) {
  union { v16bf v; v8bf h[2]; } u;
  u.h[0] = *(const v8bf*)(p);
  u.h[1] = *(const v8bf*)(p + 8);
  return u.v;
}

// A-matrix 16-bit fragment: lanes 0-15 hold K 0..15, lanes 16-31 hold
// K 16..31 (contiguous, matching ld16 B-fragments). `row` points at K=0 of this lane's M row.
__device__ __forceinline__ v16bf ld_afrag(const bf16* row, int half) {
  union { v16bf v; v8bf h[2]; } u;
  u.h[0] = *(const v8bf*)(row + half * 16);
  u.h[1] = *(const v8bf*)(row + half * 16 + 8);
  return u.v;
}

__device__ __forceinline__ void wait_async_lds() {
#if HAVE_ASYNC_LDS
#if __has_builtin(__builtin_amdgcn_s_wait_asynccnt)
  __builtin_amdgcn_s_wait_asynccnt(0);
#else
  asm volatile("s_wait_asynccnt 0" ::: "memory");
#endif
#endif
}

// ---------------------------------------------------------------------------
// Projection GEMM: O[M,4096] = W[M,C] @ In[b][C,4096], fp32 in -> bf16 out.
// Workgroup tile 128x128, K-step 32, wave tile 16x128 (8 WMMA per K-step).
// mode 0: rows 0..255 -> q (scaled), 256..511 -> ksum, 512..767 -> v
// mode 1: rows 0..255 -> ksum += k2   (RMW; k1 already written by mode 0 pass)
// ---------------------------------------------------------------------------
__global__ __launch_bounds__(256) void proj_gemm_kernel(
    const float* __restrict__ W, const float* __restrict__ In,
    bf16* __restrict__ q_ws, bf16* __restrict__ k_ws, bf16* __restrict__ v_ws,
    int C, int mode)
{
  __shared__ bf16 Wlds[128][40];   // [row][k]  (+8 pad: 16B-aligned rows, fewer bank conflicts)
  __shared__ bf16 InT[128][40];    // [n][k]    transposed so B-fragments read contiguous K

  const int b    = blockIdx.z;
  const int mBlk = blockIdx.y;
  const int nBlk = blockIdx.x;
  const int tid  = threadIdx.x;
  const int wave = tid >> 5, lane = tid & 31;
  const int half = lane >> 4, l16 = lane & 15;

  const float* Inb  = In + (size_t)b * C * N_SPATIAL;
  const int rowBase = mBlk * 128;
  const int nBase   = nBlk * 128;

  v8f acc[8] = {};

  const int kSteps = C / 32;
  for (int kt = 0; kt < kSteps; ++kt) {
    const int kBase = kt * 32;
    __syncthreads();
    // stage 128x32 weight tile (1024 float4 across 256 threads), cvt fp32->bf16
    #pragma unroll
    for (int i = 0; i < 4; ++i) {
      int idx = tid + i * 256;
      int row = idx >> 3, kc4 = idx & 7;
      float4 w4 = *(const float4*)(W + (size_t)(rowBase + row) * C + kBase + kc4 * 4);
      bf16* d = &Wlds[row][kc4 * 4];
      d[0] = (bf16)w4.x; d[1] = (bf16)w4.y; d[2] = (bf16)w4.z; d[3] = (bf16)w4.w;
    }
    // stage 32x128 input tile, transposed into InT[n][k]
    #pragma unroll
    for (int i = 0; i < 4; ++i) {
      int idx = tid + i * 256;
      int c = idx >> 5, n4 = idx & 31;
      float4 x4 = *(const float4*)(Inb + (size_t)(kBase + c) * N_SPATIAL + nBase + n4 * 4);
      InT[n4 * 4 + 0][c] = (bf16)x4.x;
      InT[n4 * 4 + 1][c] = (bf16)x4.y;
      InT[n4 * 4 + 2][c] = (bf16)x4.z;
      InT[n4 * 4 + 3][c] = (bf16)x4.w;
    }
    __syncthreads();

    v16bf afrag = ld_afrag(&Wlds[wave * 16 + l16][0], half);
    // batch 4 B-fragment loads, then 4 back-to-back WMMAs (one dscnt wait covers all)
    #pragma unroll
    for (int g = 0; g < 2; ++g) {
      v16bf bfr[4];
      #pragma unroll
      for (int j = 0; j < 4; ++j)
        bfr[j] = ld16(&InT[(g * 4 + j) * 16 + l16][half * 16]);
      #pragma unroll
      for (int j = 0; j < 4; ++j)
        acc[g * 4 + j] = __builtin_amdgcn_wmma_f32_16x16x32_bf16(
            false, afrag, false, bfr[j], (short)0, acc[g * 4 + j], false, false);
    }
  }

  // Epilogue: C/D layout -> lane n = l16, VGPR r = channel row r + 8*half.
  // Per lane & tile: 8 consecutive channel values -> one 16B bf16x8 store.
  const int g0 = rowBase + wave * 16;
  if (mode == 0) {
    const int which = g0 >> 8;             // 0=q, 1=k1, 2=v
    const int hd    = g0 & 255;
    const int h     = hd >> 7;
    const int d0    = (hd & 127) + half * 8;
    bf16* base = (which == 0) ? q_ws : (which == 1) ? k_ws : v_ws;
    const float s = (which == 0) ? 0.08838834764831845f : 1.0f;  // 128^-0.5 folded into q
    #pragma unroll
    for (int t = 0; t < 8; ++t) {
      int n = nBase + t * 16 + l16;
      BF8Pack pk;
      #pragma unroll
      for (int r = 0; r < 8; ++r) pk.b[r] = (bf16)(acc[t][r] * s);
      *(uint4*)(base + ((size_t)(b * 2 + h) * N_SPATIAL + n) * DHEAD + d0) = pk.u;
    }
  } else {
    const int h  = g0 >> 7;
    const int d0 = (g0 & 127) + half * 8;
    #pragma unroll
    for (int t = 0; t < 8; ++t) {
      int n = nBase + t * 16 + l16;
      uint4* dst = (uint4*)(k_ws + ((size_t)(b * 2 + h) * N_SPATIAL + n) * DHEAD + d0);
      BF8Pack pk; pk.u = *dst;                       // k1 written by mode-0 kernel
      #pragma unroll
      for (int r = 0; r < 8; ++r) pk.b[r] = (bf16)((float)pk.b[r] + acc[t][r]);
      *dst = pk.u;
    }
  }
}

// ---------------------------------------------------------------------------
// Flash attention, one (b,h) per blockIdx.y, 128 query rows per workgroup.
// Per wave: 16x128 output tile; key blocks of 32; 16 WMMAs / wave / key block.
// K block is staged with GLOBAL_LOAD_ASYNC_TO_LDS_B128 (ASYNCcnt) when the
// builtin is available; V is transposed in flight so it stays on the VGPR path.
// ---------------------------------------------------------------------------
__global__ __launch_bounds__(256) void attn_kernel(
    const bf16* __restrict__ q_ws, const bf16* __restrict__ k_ws,
    const bf16* __restrict__ v_ws, float* __restrict__ out)
{
  __shared__ bf16 Klds[32][136];     // [key][d]  (B-frags for S read contiguous d)
  __shared__ bf16 Vt[128][40];       // [d][key]  (B-frags for PV read contiguous key)
  __shared__ bf16 Plds[8][16][40];   // per-wave P tile for C-layout -> A-layout shuffle

  const int bh   = blockIdx.y;
  const int mBlk = blockIdx.x;
  const int tid  = threadIdx.x;
  const int wave = tid >> 5, lane = tid & 31;
  const int half = lane >> 4, l16 = lane & 15;

  const bf16* Q = q_ws + (size_t)bh * N_SPATIAL * DHEAD;
  const bf16* K = k_ws + (size_t)bh * N_SPATIAL * DHEAD;
  const bf16* V = v_ws + (size_t)bh * N_SPATIAL * DHEAD;

  // Q A-fragments: 16 query rows x d=128 (4 chunks of K=32), resident all loop long.
  const int qBase = mBlk * 128 + wave * 16;
  v16bf qf[4];
  {
    const bf16* qrow = Q + (size_t)(qBase + l16) * DHEAD;
    #pragma unroll
    for (int dc = 0; dc < 4; ++dc) qf[dc] = ld_afrag(qrow + dc * 32, half);
  }

  v8f o[8] = {};
  float mrow[8], lrow[8];
  #pragma unroll
  for (int r = 0; r < 8; ++r) { mrow[r] = -3.0e38f; lrow[r] = 0.0f; }

  for (int kb = 0; kb < N_SPATIAL / 32; ++kb) {
    const bf16* Kblk = K + (size_t)kb * 32 * DHEAD;
    const bf16* Vblk = V + (size_t)kb * 32 * DHEAD;
    __syncthreads();
    // stage K block (row-major, async DMA when available) and V block (transposed)
    #pragma unroll
    for (int i = 0; i < 2; ++i) {
      int idx = tid + i * 256;               // uint4 (8 bf16) index, 512 total
      int key = idx >> 4, d8 = idx & 15;
#if HAVE_ASYNC_LDS
      __builtin_amdgcn_global_load_async_to_lds_b128(
          (as1_v4i)(Kblk + key * DHEAD + d8 * 8),
          (as3_v4i)(&Klds[key][d8 * 8]), 0, 0);
#else
      *(uint4*)(&Klds[key][d8 * 8]) = *(const uint4*)(Kblk + key * DHEAD + d8 * 8);
#endif
      BF8Pack pv; pv.u = *(const uint4*)(Vblk + key * DHEAD + d8 * 8);
      #pragma unroll
      for (int e = 0; e < 8; ++e) Vt[d8 * 8 + e][key] = pv.b[e];
    }
    if (kb + 1 < N_SPATIAL / 32) {           // global_prefetch_b8 next K/V block
      __builtin_prefetch(Kblk + 32 * DHEAD + tid * 16, 0, 1);
      __builtin_prefetch(Vblk + 32 * DHEAD + tid * 16, 0, 1);
    }
    wait_async_lds();
    __syncthreads();

    // S = (q*scale) @ ksum^T : two 16x16 f32 tiles (keys 0-15, 16-31)
    v8f s0 = {}, s1 = {};
    #pragma unroll
    for (int g = 0; g < 2; ++g) {            // batch 4 loads, then 4 WMMAs
      v16bf bk[4];
      #pragma unroll
      for (int j = 0; j < 2; ++j) {
        int dc = g * 2 + j;
        bk[2 * j]     = ld16(&Klds[l16][dc * 32 + half * 16]);
        bk[2 * j + 1] = ld16(&Klds[16 + l16][dc * 32 + half * 16]);
      }
      #pragma unroll
      for (int j = 0; j < 2; ++j) {
        s0 = __builtin_amdgcn_wmma_f32_16x16x32_bf16(false, qf[g * 2 + j], false, bk[2 * j],     (short)0, s0, false, false);
        s1 = __builtin_amdgcn_wmma_f32_16x16x32_bf16(false, qf[g * 2 + j], false, bk[2 * j + 1], (short)0, s1, false, false);
      }
    }

    // online softmax: row = VGPR index r (+8*half), reduce over the 16 lanes of a half
    #pragma unroll
    for (int r = 0; r < 8; ++r) {
      float vmx = fmaxf(s0[r], s1[r]);
      #pragma unroll
      for (int off = 1; off < 16; off <<= 1) vmx = fmaxf(vmx, __shfl_xor(vmx, off, 32));
      float mnew  = fmaxf(mrow[r], vmx);
      float alpha = __expf(mrow[r] - mnew);
      mrow[r] = mnew;
      float p0 = __expf(s0[r] - mnew);
      float p1 = __expf(s1[r] - mnew);
      float ps = p0 + p1;
      #pragma unroll
      for (int off = 1; off < 16; off <<= 1) ps += __shfl_xor(ps, off, 32);
      lrow[r] = lrow[r] * alpha + ps;
      #pragma unroll
      for (int t = 0; t < 8; ++t) o[t][r] *= alpha;
      Plds[wave][half * 8 + r][l16]      = (bf16)p0;   // C-layout -> LDS
      Plds[wave][half * 8 + r][16 + l16] = (bf16)p1;
    }
    __syncthreads();

    // O += P @ V  (P re-read in A-layout from LDS); batch 4 loads then 4 WMMAs
    v16bf pf = ld_afrag(&Plds[wave][l16][0], half);
    #pragma unroll
    for (int g = 0; g < 2; ++g) {
      v16bf bv[4];
      #pragma unroll
      for (int j = 0; j < 4; ++j)
        bv[j] = ld16(&Vt[(g * 4 + j) * 16 + l16][half * 16]);
      #pragma unroll
      for (int j = 0; j < 4; ++j)
        o[g * 4 + j] = __builtin_amdgcn_wmma_f32_16x16x32_bf16(
            false, pf, false, bv[j], (short)0, o[g * 4 + j], false, false);
    }
  }

  // Final normalize + store to out[b, h*128+d, n]: lane = d, rows r -> consecutive
  // queries -> two contiguous float4 stores per lane per d-tile.
  float inv[8];
  #pragma unroll
  for (int r = 0; r < 8; ++r) inv[r] = 1.0f / lrow[r];
  #pragma unroll
  for (int t = 0; t < 8; ++t) {
    float* dst = out + ((size_t)bh * DHEAD + t * 16 + l16) * N_SPATIAL
                     + (size_t)mBlk * 128 + wave * 16 + half * 8;
    float4 lo = { o[t][0] * inv[0], o[t][1] * inv[1], o[t][2] * inv[2], o[t][3] * inv[3] };
    float4 hi = { o[t][4] * inv[4], o[t][5] * inv[5], o[t][6] * inv[6], o[t][7] * inv[7] };
    *(float4*)dst       = lo;
    *(float4*)(dst + 4) = hi;
  }
}

// ---------------------------------------------------------------------------
extern "C" void kernel_launch(void* const* d_in, const int* in_sizes, int n_in,
                              void* d_out, int out_size, void* d_ws, size_t ws_size,
                              hipStream_t stream) {
  (void)in_sizes; (void)n_in; (void)out_size; (void)ws_size;
  const float* fmap = (const float*)d_in[0];   // [2,256,64,64]
  const float* x    = (const float*)d_in[1];   // [2,2048,64,64]
  const float* Wqkv = (const float*)d_in[2];   // [768,256]
  const float* Wk2  = (const float*)d_in[3];   // [256,2048]
  float* out = (float*)d_out;

  const size_t tElems = (size_t)BH * N_SPATIAL * DHEAD;  // 2,097,152 bf16 per tensor
  bf16* q_ws = (bf16*)d_ws;
  bf16* k_ws = q_ws + tElems;
  bf16* v_ws = k_ws + tElems;

  dim3 blk(256);
  // qkv projection: 768 rows x 4096 cols, K=256
  proj_gemm_kernel<<<dim3(32, 6, 2), blk, 0, stream>>>(Wqkv, fmap, q_ws, k_ws, v_ws, 256, 0);
  // k2 projection + fold into ksum: 256 rows x 4096 cols, K=2048
  proj_gemm_kernel<<<dim3(32, 2, 2), blk, 0, stream>>>(Wk2, x, q_ws, k_ws, v_ws, 2048, 1);
  // flash attention: 4 (b,h) x 32 query blocks
  attn_kernel<<<dim3(32, 4), blk, 0, stream>>>(q_ws, k_ws, v_ws, out);
}
